// SimpleHGNConv_87514253623575
// MI455X (gfx1250) — matrix operand order repaired and checked by reading, not compiled, not run on hardware
//
#include <hip/hip_runtime.h>
#include <hip/hip_bf16.h>

#define N_NODES 10000
#define N_EDGES 320000
#define IN_DIM  256
#define D_DIM   64
#define H_HEADS 4
#define ED_DIM  32
#define T_TYPES 8
#define OUT_COLS 256   /* D*H */

typedef __attribute__((ext_vector_type(8)))  __bf16 v8bf;
typedef __attribute__((ext_vector_type(16))) __bf16 v16bf;
typedef __attribute__((ext_vector_type(8)))  float  v8f;

// ---------------- round-to-nearest-even f32 -> bf16 (stored as ushort) ------
__device__ inline unsigned short f32_to_bf16_rne(float x) {
    unsigned u = __float_as_uint(x);
    if ((u & 0x7FFFFFFFu) > 0x7F800000u) return (unsigned short)0x7FC0u; // NaN
    u += 0x7FFFu + ((u >> 16) & 1u);
    return (unsigned short)(u >> 16);
}

// ---------------- prep: feat -> bf16 ----------------------------------------
__global__ __launch_bounds__(256) void cvt_feat_kernel(
    const float* __restrict__ f, unsigned short* __restrict__ o, int total) {
    int i = blockIdx.x * 256 + threadIdx.x;
    if (i < total) o[i] = f32_to_bf16_rne(f[i]);
}

// ---------------- prep: W[k][n] -> Wt[n][k] bf16 -----------------------------
__global__ __launch_bounds__(256) void transpose_w_kernel(
    const float* __restrict__ w, unsigned short* __restrict__ o) {
    int k = blockIdx.x;      // 0..255
    int n = threadIdx.x;     // 0..255
    o[(size_t)n * 256 + k] = f32_to_bf16_rne(w[(size_t)k * 256 + n]);
}

// ---------------- prep: he_type[t][h] (typed-linear fully contracted) --------
__global__ void he_kernel(const float* __restrict__ ee,   // [T,ED]
                          const float* __restrict__ W_r,  // [T,ED,ED*H]
                          const float* __restrict__ a_e,  // [H,ED]
                          float* __restrict__ he) {       // [T,H]
    int tid = threadIdx.x;           // 32 threads
    int t = tid >> 2, h = tid & 3;
    float acc = 0.f;
    for (int o = 0; o < ED_DIM; ++o) {
        float to = 0.f;
        for (int i = 0; i < ED_DIM; ++i)
            to += ee[t * ED_DIM + i] * W_r[((size_t)t * ED_DIM + i) * 128 + h * ED_DIM + o];
        acc += to * a_e[h * ED_DIM + o];
    }
    he[t * H_HEADS + h] = acc;
}

// ---------------- main WMMA GEMM: emb = feat@W, res = feat@W_res + b_res -----
// One wave computes a 16x64 tile (4 x 16x16 sub-tiles sharing the A fragment).
// 625 M-tiles x 8 N-groups (groups 0..3 -> W/emb, 4..7 -> W_res/res).
// K=256 -> 8 K-steps x 4 wmma = 32 v_wmma per wave, 2.5 b128 loads per wmma.
__global__ __launch_bounds__(256) void gemm_wmma_kernel(
    const __bf16* __restrict__ Abf,   // [10000,256] bf16
    const __bf16* __restrict__ BtW,   // [256 n][256 k] bf16 (transposed W)
    const __bf16* __restrict__ BtWr,  // transposed W_res
    const float* __restrict__ b_res,  // [256]
    float* __restrict__ emb,          // [10000,256]
    float* __restrict__ res) {        // [10000,256]
    const int lane = threadIdx.x & 31;
    const int wave = threadIdx.x >> 5;
    const int flat = blockIdx.x * 8 + wave;   // 0..4999
    const int m_tile = flat >> 3;             // 0..624
    const int n_grp  = flat & 7;              // 0..7
    const bool isRes = n_grp >= 4;
    const int c0 = (n_grp & 3) * 64;          // first output column of group
    const int m0 = m_tile * 16;
    const __bf16* __restrict__ Bt = isRes ? BtWr : BtW;

    const int ml   = lane & 15;    // A row / B column within 16x16 sub-tile
    const int half = lane >> 4;    // lane half selects K sub-range

    // A: lane half 0 -> K = {0..7, 16..23}; half 1 -> K = {8..15, 24..31}
    const __bf16* aptr = Abf + (size_t)(m0 + ml) * IN_DIM + half * 8;
    // B: lane half 0 -> K = {0..7, 16..23} of column n; half 1 -> K = {8..15, 24..31}
    const __bf16* bptr0 = Bt + (size_t)(c0 + ml) * IN_DIM + half * 8;

    v8f c0v = {}, c1v = {}, c2v = {}, c3v = {};
    #pragma unroll
    for (int k0 = 0; k0 < IN_DIM; k0 += 32) {
        v8bf a0 = *(const v8bf*)(aptr + k0);
        v8bf a1 = *(const v8bf*)(aptr + k0 + 16);
        v16bf av = __builtin_shufflevector(a0, a1, 0,1,2,3,4,5,6,7,8,9,10,11,12,13,14,15);
        #pragma unroll
        for (int j = 0; j < 4; ++j) {
            const __bf16* bp = bptr0 + (size_t)(j * 16) * IN_DIM + k0;
            v8bf b0 = *(const v8bf*)(bp);
            v8bf b1 = *(const v8bf*)(bp + 16);
            v16bf bv = __builtin_shufflevector(b0, b1, 0,1,2,3,4,5,6,7,8,9,10,11,12,13,14,15);
            v8f& acc = (j == 0) ? c0v : (j == 1) ? c1v : (j == 2) ? c2v : c3v;
            acc = __builtin_amdgcn_wmma_f32_16x16x32_bf16(
                      /*neg_a=*/false, av, /*neg_b=*/false, bv,
                      /*c_mod=*/(short)0, acc, /*reuse_a=*/false, /*reuse_b=*/false);
        }
    }

    float* __restrict__ Out = isRes ? res : emb;
    #pragma unroll
    for (int j = 0; j < 4; ++j) {
        const v8f& acc = (j == 0) ? c0v : (j == 1) ? c1v : (j == 2) ? c2v : c3v;
        const int ncol = c0 + j * 16 + ml;
        const float bias = isRes ? b_res[ncol] : 0.f;
        #pragma unroll
        for (int r = 0; r < 8; ++r) {
            int mrow = m0 + r + half * 8;   // C/D layout: VGPR r = rows r / r+8
            float v = acc[r];
            if (isRes) v += bias;
            else       v = (v != v) ? 0.f : v;   // nan_to_num on emb
            Out[(size_t)mrow * OUT_COLS + ncol] = v;
        }
    }
}

// ---------------- per-node attention dots hl, hr -----------------------------
__global__ __launch_bounds__(256) void node_attn_kernel(
    const float* __restrict__ emb, const float* __restrict__ a_l,
    const float* __restrict__ a_r, float* __restrict__ hl, float* __restrict__ hr) {
    __shared__ float sl[H_HEADS], sr[H_HEADS];
    const int n = blockIdx.x;
    const int t = threadIdx.x;            // 0..255 == h*64+d
    if (t < H_HEADS) { sl[t] = 0.f; sr[t] = 0.f; }
    __syncthreads();
    float v  = emb[(size_t)n * OUT_COLS + t];
    float wl = v * a_l[t];
    float wr = v * a_r[t];
    #pragma unroll
    for (int off = 16; off > 0; off >>= 1) {
        wl += __shfl_down(wl, off, 32);
        wr += __shfl_down(wr, off, 32);
    }
    if ((t & 31) == 0) {
        int h = t >> 6;
        atomicAdd(&sl[h], wl);   // ds_add_f32
        atomicAdd(&sr[h], wr);
    }
    __syncthreads();
    if (t < H_HEADS) {
        hl[n * H_HEADS + t] = sl[t];
        hr[n * H_HEADS + t] = sr[t];
    }
}

// ---------------- edge gather + attention + scatter-add ----------------------
// One block handles 4 edges; thread t = h*64+d. Scatter directly into [N, D, H]
// layout (index c*256 + d*4 + h) so the epilogue is streaming.
__global__ __launch_bounds__(256) void edge_kernel(
    const float* __restrict__ emb, const float* __restrict__ hl,
    const float* __restrict__ hr,  const float* __restrict__ he,
    const float* __restrict__ ew,  const int* __restrict__ row,
    const int* __restrict__ col,   const int* __restrict__ et,
    float* __restrict__ agg) {
    const int e0 = blockIdx.x * 4;
    const int t  = threadIdx.x;
    const int h  = t >> 6, d = t & 63;
    #pragma unroll
    for (int k = 0; k < 4; ++k) {
        const int e  = e0 + k;
        const int r  = row[e];
        const int c  = col[e];
        const int ty = et[e];
        float z = hl[r * H_HEADS + h] + hr[c * H_HEADS + h] + he[ty * H_HEADS + h];
        z = (z > 0.f) ? z : z * 0.2f;                       // leaky_relu
        float att = (1.f / (1.f + __expf(-z))) * ew[e];     // sigmoid * edge_weight
        float val = emb[(size_t)r * OUT_COLS + t] * att;
        __hip_atomic_fetch_add(&agg[(size_t)c * OUT_COLS + d * H_HEADS + h], val,
                               __ATOMIC_RELAXED, __HIP_MEMORY_SCOPE_AGENT);
    }
}

// ---------------- epilogue: out = elu(agg + res) -----------------------------
__global__ __launch_bounds__(256) void final_kernel(
    const float* __restrict__ agg, const float* __restrict__ res,
    float* __restrict__ out, int total) {
    int i = blockIdx.x * 256 + threadIdx.x;
    if (i < total) {
        float x = agg[i] + res[i];
        out[i] = (x > 0.f) ? x : (__expf(x) - 1.f);
    }
}

extern "C" void kernel_launch(void* const* d_in, const int* in_sizes, int n_in,
                              void* d_out, int out_size, void* d_ws, size_t ws_size,
                              hipStream_t stream) {
    (void)in_sizes; (void)n_in; (void)out_size; (void)ws_size;
    const float* feat    = (const float*)d_in[0];
    const float* ew      = (const float*)d_in[1];
    const int*   row     = (const int*)  d_in[2];
    const int*   col     = (const int*)  d_in[3];
    const int*   etype   = (const int*)  d_in[4];
    const float* W       = (const float*)d_in[5];
    const float* ee_tab  = (const float*)d_in[6];
    const float* W_r     = (const float*)d_in[7];
    const float* a_l     = (const float*)d_in[8];
    const float* a_r     = (const float*)d_in[9];
    const float* a_e     = (const float*)d_in[10];
    const float* W_res   = (const float*)d_in[11];
    const float* b_res   = (const float*)d_in[12];
    float* out = (float*)d_out;

    // workspace layout (256B aligned)
    char* base = (char*)d_ws;
    size_t off = 0;
    auto take = [&](size_t bytes) { char* p = base + off;
                                    off = (off + bytes + 255) & ~(size_t)255; return p; };
    unsigned short* featbf = (unsigned short*)take((size_t)N_NODES * IN_DIM * 2);
    unsigned short* wt     = (unsigned short*)take((size_t)IN_DIM * OUT_COLS * 2);
    unsigned short* wrt    = (unsigned short*)take((size_t)IN_DIM * OUT_COLS * 2);
    float* emb = (float*)take((size_t)N_NODES * OUT_COLS * 4);
    float* res = (float*)take((size_t)N_NODES * OUT_COLS * 4);
    float* hl  = (float*)take((size_t)N_NODES * H_HEADS * 4);
    float* hr  = (float*)take((size_t)N_NODES * H_HEADS * 4);
    float* he  = (float*)take((size_t)T_TYPES * H_HEADS * 4);
    float* agg = (float*)take((size_t)N_NODES * OUT_COLS * 4);

    const int featTotal = N_NODES * IN_DIM;

    // prep
    hipMemsetAsync(agg, 0, (size_t)N_NODES * OUT_COLS * 4, stream);
    cvt_feat_kernel<<<(featTotal + 255) / 256, 256, 0, stream>>>(feat, featbf, featTotal);
    transpose_w_kernel<<<IN_DIM, 256, 0, stream>>>(W, wt);
    transpose_w_kernel<<<IN_DIM, 256, 0, stream>>>(W_res, wrt);
    he_kernel<<<1, 32, 0, stream>>>(ee_tab, W_r, a_e, he);

    // GEMMs (WMMA bf16): 625 M-tiles * 8 N-groups = 5000 waves / 8 per block
    gemm_wmma_kernel<<<625, 256, 0, stream>>>(
        (const __bf16*)featbf, (const __bf16*)wt, (const __bf16*)wrt,
        b_res, emb, res);

    // per-node head dots
    node_attn_kernel<<<N_NODES, 256, 0, stream>>>(emb, a_l, a_r, hl, hr);

    // edge message passing (L2-resident gather + f32 atomic scatter)
    edge_kernel<<<N_EDGES / 4, 256, 0, stream>>>(emb, hl, hr, he, ew, row, col, etype, agg);

    // epilogue
    final_kernel<<<(featTotal + 255) / 256, 256, 0, stream>>>(agg, res, out, featTotal);
}
